// Rope_MHA_19018115187101
// MI455X (gfx1250) — matrix operand order repaired and checked by reading, not compiled, not run on hardware
//
#include <hip/hip_runtime.h>

#define B_  4
#define S_  2048
#define D_  2048
#define H_  16
#define DH_ 128

typedef __bf16 bf16_t;
typedef bf16_t v16bf __attribute__((ext_vector_type(16)));
typedef bf16_t v8bf  __attribute__((ext_vector_type(8)));
typedef float  v8f   __attribute__((ext_vector_type(8)));
typedef unsigned int v4u __attribute__((ext_vector_type(4)));
typedef int v4i __attribute__((ext_vector_type(4)));
typedef int v8i __attribute__((ext_vector_type(8)));

union AFrag { v16bf v; v8bf h[2]; };

static __device__ __forceinline__ v8f wmma_bf16(v16bf a, v16bf b, v8f c) {
  // D = A(16x32 bf16) * B(32x16 bf16) + C(16x16 f32)
  return __builtin_amdgcn_wmma_f32_16x16x32_bf16(
      false, a, false, b, (short)0, c, false, false);
}

#if __has_builtin(__builtin_amdgcn_tensor_load_to_lds) && \
    __has_builtin(__builtin_amdgcn_s_wait_tensorcnt)
#define HAVE_TDM 1
#else
#define HAVE_TDM 0
#endif

#if HAVE_TDM
// TDM: DMA a 2D tile [tile_dim1=128 lines x tile_dim0=32 elems, 2B/elem]
// from global (line stride 2048 elems) into LDS at byte offset lds_off.
// D# layout per CDNA5 ISA.
static __device__ __forceinline__ void tdm_load_b_panel(
    unsigned lds_off, unsigned long long gaddr, unsigned tensor_dim1) {
  v4u g0;
  g0[0] = 1u;                                   // count=1, user descriptor
  g0[1] = lds_off;                              // lds_addr (bytes)
  g0[2] = (unsigned)(gaddr & 0xffffffffu);      // global_addr[31:0]
  g0[3] = (unsigned)((gaddr >> 32) & 0x01ffffffu) | 0x80000000u;  // addr hi | type=2
  v8i g1;
  g1[0] = (int)(1u << 16);                      // data_size=1 (2 bytes)
  g1[1] = (int)(2048u << 16);                   // tensor_dim0 = 2048 (K extent)
  g1[2] = (int)((tensor_dim1 & 0xffffu) << 16); // tensor_dim1 (lines in tensor)
  g1[3] = (int)(32u << 16);                     // tile_dim0 = 32
  g1[4] = 128;                                  // tile_dim1 = 128 lines
  g1[5] = 2048;                                 // tensor_dim0_stride = 2048 elems
  g1[6] = 0;
  g1[7] = 0;
  v4i gz = {0, 0, 0, 0};
#if defined(__clang_major__) && (__clang_major__ >= 23)
  v8i z8 = {0, 0, 0, 0, 0, 0, 0, 0};
  __builtin_amdgcn_tensor_load_to_lds(g0, g1, gz, gz, z8, 0);
#else
  __builtin_amdgcn_tensor_load_to_lds(g0, g1, gz, gz, 0);
#endif
}
#endif

// ---------------------------------------------------------------------------
// Kernel 1: fp32 -> bf16 conversion (memory-bound prologue)
// ---------------------------------------------------------------------------
__global__ void cvt_f32_bf16(const float* __restrict__ in,
                             bf16_t* __restrict__ out, int n) {
  int i = blockIdx.x * blockDim.x + threadIdx.x;
  if (i < n) out[i] = (bf16_t)in[i];
}

// ---------------------------------------------------------------------------
// Kernel 2: QKV = x @ qkv^T with fused RoPE.
//   Workgroup = 8 waves sharing a TDM-staged 128col x 32K B-panel in LDS
//   (double buffered). Each wave owns one 16-row M tile and a full head
//   (128 cols = 8 WMMA accumulators), so rotate_half pairs (t, t+4) are
//   wave-internal. blockIdx.x: 48 tasks = {Q,K,V} x 16 heads.
// ---------------------------------------------------------------------------
__global__ __launch_bounds__(256) void qkv_rope_gemm(
    const bf16_t* __restrict__ xb, const bf16_t* __restrict__ wb,
    bf16_t* __restrict__ qO, bf16_t* __restrict__ kO, bf16_t* __restrict__ vTO,
    float* __restrict__ khOut, float* __restrict__ vhOut) {
  __shared__ __align__(16) bf16_t Bp[2 * 4096];  // 2 x 8KB panels, [col][k0..31]
  const int tid = threadIdx.x;
  const int lane = tid & 31, wid = tid >> 5;
  const int ml = lane & 15, hi = lane >> 4;
  const int task = blockIdx.x;            // 0..47
  const int region = task >> 4;           // 0=Q, 1=K, 2=V
  const int head = task & 15;
  const int colBase = region * D_ + head * DH_;
  const int mt = blockIdx.y * 8 + wid;    // 0..511
  const bf16_t* arow = xb + (size_t)(mt * 16 + ml) * D_;

  v8f acc[8] = {};

#if HAVE_TDM
  const unsigned long long gbase =
      (unsigned long long)(const void*)(wb + (size_t)colBase * D_);
  const unsigned lds0 = (unsigned)(size_t)(void*)&Bp[0];
  if (tid == 0) tdm_load_b_panel(lds0, gbase, 3 * D_);
  for (int i = 0; i < D_ / 32; ++i) {
    const int kk = i * 32;
    if (tid == 0) {
      if (i + 1 < D_ / 32) {
        tdm_load_b_panel(lds0 + ((unsigned)((i + 1) & 1)) * 8192u,
                         gbase + (unsigned long long)(kk + 32) * 2u, 3 * D_);
        __builtin_amdgcn_s_wait_tensorcnt(1);  // current panel done
      } else {
        __builtin_amdgcn_s_wait_tensorcnt(0);
      }
    }
    __syncthreads();
    const bf16_t* Bcur = &Bp[(i & 1) * 4096];
    AFrag a;
    a.h[0] = *(const v8bf*)(arow + kk + hi * 16);
    a.h[1] = *(const v8bf*)(arow + kk + 8 + hi * 16);
#pragma unroll
    for (int t = 0; t < 8; ++t) {
      v16bf b = *(const v16bf*)&Bcur[(t * 16 + ml) * 32 + hi * 16];
      acc[t] = wmma_bf16(a.v, b, acc[t]);
    }
    __syncthreads();  // WAR: panel reused two iterations later
  }
#else
  for (int i = 0; i < D_ / 32; ++i) {
    const int kk = i * 32;
    // cooperative sync staging: wave w stages columns w*16..w*16+15
    *(v16bf*)&Bp[(wid * 16 + ml) * 32 + hi * 16] =
        *(const v16bf*)(wb + (size_t)(colBase + wid * 16 + ml) * D_ + kk + hi * 16);
    __syncthreads();
    AFrag a;
    a.h[0] = *(const v8bf*)(arow + kk + hi * 16);
    a.h[1] = *(const v8bf*)(arow + kk + 8 + hi * 16);
#pragma unroll
    for (int t = 0; t < 8; ++t) {
      v16bf b = *(const v16bf*)&Bp[(t * 16 + ml) * 32 + hi * 16];
      acc[t] = wmma_bf16(a.v, b, acc[t]);
    }
    __syncthreads();
  }
#endif

  if (region == 2) {
    // ---- V: fp32 vh to d_out + bf16 transposed [dh][s] for the PV matmul ----
#pragma unroll
    for (int t = 0; t < 8; ++t) {
      const int dh = t * 16 + ml;
#pragma unroll
      for (int r = 0; r < 8; ++r) {
        const int grow = mt * 16 + r + 8 * hi;
        const int bb = grow >> 11, sp = grow & (S_ - 1);
        const float v = acc[t][r];
        vhOut[(((size_t)bb * H_ + head) * S_ + sp) * DH_ + dh] = v;
        vTO[(((size_t)bb * H_ + head) * DH_ + dh) * S_ + sp] = (bf16_t)v;
      }
    }
  } else {
    const bool isQ = (region == 0);
#pragma unroll
    for (int t = 0; t < 4; ++t) {
      const int j = t * 16 + ml;  // RoPE frequency index 0..63
      const float invf = __powf(10000.f, -(float)j * (1.0f / 64.0f));
#pragma unroll
      for (int r = 0; r < 8; ++r) {
        const int grow = mt * 16 + r + 8 * hi;
        const int bb = grow >> 11, sp = grow & (S_ - 1);
        const float ang = (float)sp * invf;
        const float cs = __cosf(ang), sn = __sinf(ang);
        const float t1 = acc[t][r], t2 = acc[t + 4][r];
        const float r0 = t1 * cs - t2 * sn;   // first half of head dim
        const float r1 = t2 * cs + t1 * sn;   // second half
        const size_t base = (((size_t)bb * H_ + head) * S_ + sp) * DH_ + j;
        if (isQ) {
          qO[base] = (bf16_t)r0;  qO[base + 64] = (bf16_t)r1;
        } else {
          kO[base] = (bf16_t)r0;  kO[base + 64] = (bf16_t)r1;
          khOut[base] = r0;       khOut[base + 64] = r1;
        }
      }
    }
  }
}

// ---------------------------------------------------------------------------
// Kernel 3: causal flash attention, one wave per 16-row Q tile.
//   32-key blocks: 8 WMMAs for scores, online softmax (wave32 shfl_xor row
//   reductions over the 16-lane C-layout halves), P staged through LDS
//   (C-layout -> A-fragment layout), 8 WMMAs for P@V (V pre-transposed).
// ---------------------------------------------------------------------------
__global__ __launch_bounds__(32) void attn_fwd(
    const bf16_t* __restrict__ q, const bf16_t* __restrict__ k,
    const bf16_t* __restrict__ vT, bf16_t* __restrict__ o) {
  __shared__ __align__(16) bf16_t pls[16 * 32];
  const int lane = threadIdx.x;
  const int ml = lane & 15, hi = lane >> 4;
  const int id = blockIdx.x;           // 0..8191
  const int qt = id & 127;
  const int h  = (id >> 7) & 15;
  const int bb = id >> 11;
  const int q0 = qt * 16;

  const bf16_t* qb = q  + ((size_t)bb * H_ + h) * (size_t)S_ * DH_;
  const bf16_t* kb = k  + ((size_t)bb * H_ + h) * (size_t)S_ * DH_;
  const bf16_t* vb = vT + ((size_t)bb * H_ + h) * (size_t)DH_ * S_;

  AFrag qf[4];
  const bf16_t* qrow = qb + (size_t)(q0 + ml) * DH_;
#pragma unroll
  for (int d = 0; d < 4; ++d) {
    qf[d].h[0] = *(const v8bf*)(qrow + d * 32 + hi * 16);
    qf[d].h[1] = *(const v8bf*)(qrow + d * 32 + 8 + hi * 16);
  }

  v8f acc[8] = {};
  float mrow[8], lrow[8];
#pragma unroll
  for (int r = 0; r < 8; ++r) { mrow[r] = -1e30f; lrow[r] = 0.f; }
  const float scale = 0.08838834764831845f;  // 1/sqrt(128)

  for (int kblk = 0; kblk <= q0 + 15; kblk += 32) {
    v8f s0 = {}, s1 = {};
#pragma unroll
    for (int d = 0; d < 4; ++d) {
      const bf16_t* kd = kb + d * 32 + hi * 16;
      v16bf kf0 = *(const v16bf*)(kd + (size_t)(kblk + ml) * DH_);
      v16bf kf1 = *(const v16bf*)(kd + (size_t)(kblk + 16 + ml) * DH_);
      s0 = wmma_bf16(qf[d].v, kf0, s0);
      s1 = wmma_bf16(qf[d].v, kf1, s1);
    }
    const int key0 = kblk + ml, key1 = kblk + 16 + ml;
#pragma unroll
    for (int r = 0; r < 8; ++r) {
      const int qr = q0 + r + 8 * hi;
      float v0 = (key0 <= qr) ? s0[r] * scale : -1e30f;
      float v1 = (key1 <= qr) ? s1[r] * scale : -1e30f;
      float bm = fmaxf(v0, v1);
#pragma unroll
      for (int off = 1; off < 16; off <<= 1)
        bm = fmaxf(bm, __shfl_xor(bm, off, 32));
      const float mnew = fmaxf(mrow[r], bm);
      const float corr = __expf(mrow[r] - mnew);
      const float e0 = (key0 <= qr) ? __expf(v0 - mnew) : 0.f;
      const float e1 = (key1 <= qr) ? __expf(v1 - mnew) : 0.f;
      float rs = e0 + e1;
#pragma unroll
      for (int off = 1; off < 16; off <<= 1) rs += __shfl_xor(rs, off, 32);
      lrow[r] = lrow[r] * corr + rs;
      mrow[r] = mnew;
#pragma unroll
      for (int n = 0; n < 8; ++n) acc[n][r] *= corr;
      const int prow = r + 8 * hi;
      pls[prow * 32 + ml]      = (bf16_t)e0;
      pls[prow * 32 + 16 + ml] = (bf16_t)e1;
    }
    __syncthreads();  // single-wave workgroup: lowers to waitcnts (+s_nop)
    AFrag pf;
    pf.h[0] = *(const v8bf*)(&pls[ml * 32 + hi * 16]);
    pf.h[1] = *(const v8bf*)(&pls[ml * 32 + 8 + hi * 16]);
#pragma unroll
    for (int n = 0; n < 8; ++n) {
      v16bf vf = *(const v16bf*)(vb + (size_t)(n * 16 + ml) * S_ + kblk + hi * 16);
      acc[n] = wmma_bf16(pf.v, vf, acc[n]);
    }
    __syncthreads();
  }
#pragma unroll
  for (int n = 0; n < 8; ++n) {
#pragma unroll
    for (int r = 0; r < 8; ++r) {
      const int qr = q0 + r + 8 * hi;
      const float val = acc[n][r] / lrow[r];
      o[((size_t)bb * S_ + qr) * D_ + h * DH_ + n * 16 + ml] = (bf16_t)val;
    }
  }
}

// ---------------------------------------------------------------------------
// Kernel 4: out = attn_out @ wo^T, fp32 to d_out. Same TDM-staged panel
// scheme: 8 waves share a 128-col B panel, each wave does a 16x128 strip.
// ---------------------------------------------------------------------------
__global__ __launch_bounds__(256) void out_proj(
    const bf16_t* __restrict__ ob, const bf16_t* __restrict__ wob,
    float* __restrict__ out) {
  __shared__ __align__(16) bf16_t Bp[2 * 4096];
  const int tid = threadIdx.x;
  const int lane = tid & 31, wid = tid >> 5;
  const int ml = lane & 15, hi = lane >> 4;
  const int colBase = blockIdx.x * 128;   // 16 column groups
  const int mt = blockIdx.y * 8 + wid;    // 0..511
  const bf16_t* arow = ob + (size_t)(mt * 16 + ml) * D_;

  v8f acc[8] = {};

#if HAVE_TDM
  const unsigned long long gbase =
      (unsigned long long)(const void*)(wob + (size_t)colBase * D_);
  const unsigned lds0 = (unsigned)(size_t)(void*)&Bp[0];
  if (tid == 0) tdm_load_b_panel(lds0, gbase, D_);
  for (int i = 0; i < D_ / 32; ++i) {
    const int kk = i * 32;
    if (tid == 0) {
      if (i + 1 < D_ / 32) {
        tdm_load_b_panel(lds0 + ((unsigned)((i + 1) & 1)) * 8192u,
                         gbase + (unsigned long long)(kk + 32) * 2u, D_);
        __builtin_amdgcn_s_wait_tensorcnt(1);
      } else {
        __builtin_amdgcn_s_wait_tensorcnt(0);
      }
    }
    __syncthreads();
    const bf16_t* Bcur = &Bp[(i & 1) * 4096];
    AFrag a;
    a.h[0] = *(const v8bf*)(arow + kk + hi * 16);
    a.h[1] = *(const v8bf*)(arow + kk + 8 + hi * 16);
#pragma unroll
    for (int t = 0; t < 8; ++t) {
      v16bf b = *(const v16bf*)&Bcur[(t * 16 + ml) * 32 + hi * 16];
      acc[t] = wmma_bf16(a.v, b, acc[t]);
    }
    __syncthreads();
  }
#else
  for (int i = 0; i < D_ / 32; ++i) {
    const int kk = i * 32;
    *(v16bf*)&Bp[(wid * 16 + ml) * 32 + hi * 16] =
        *(const v16bf*)(wob + (size_t)(colBase + wid * 16 + ml) * D_ + kk + hi * 16);
    __syncthreads();
    AFrag a;
    a.h[0] = *(const v8bf*)(arow + kk + hi * 16);
    a.h[1] = *(const v8bf*)(arow + kk + 8 + hi * 16);
#pragma unroll
    for (int t = 0; t < 8; ++t) {
      v16bf b = *(const v16bf*)&Bp[(t * 16 + ml) * 32 + hi * 16];
      acc[t] = wmma_bf16(a.v, b, acc[t]);
    }
    __syncthreads();
  }
#endif

#pragma unroll
  for (int t = 0; t < 8; ++t) {
#pragma unroll
    for (int r = 0; r < 8; ++r)
      out[(size_t)(mt * 16 + r + 8 * hi) * D_ + colBase + t * 16 + ml] = acc[t][r];
  }
}

// ---------------------------------------------------------------------------
extern "C" void kernel_launch(void* const* d_in, const int* in_sizes, int n_in,
                              void* d_out, int out_size, void* d_ws, size_t ws_size,
                              hipStream_t stream) {
  (void)in_sizes; (void)n_in; (void)out_size; (void)ws_size;
  const float* x   = (const float*)d_in[0];   // [B,S,D]
  const float* qkv = (const float*)d_in[1];   // [3D,D]
  const float* wo  = (const float*)d_in[2];   // [D,D]

  float* out = (float*)d_out;
  const size_t nTok = (size_t)B_ * S_ * D_;   // 16,777,216
  float* khOut = out + nTok;                  // [B,H,S,DH] fp32
  float* vhOut = khOut + nTok;                // [B,H,S,DH] fp32

  // Workspace carve-out (bf16 staging), ~168 MB total (x buffer reused for o)
  char* w = (char*)d_ws;
  bf16_t* xb   = (bf16_t*)w; w += nTok * sizeof(bf16_t);
  bf16_t* qkvb = (bf16_t*)w; w += (size_t)3 * D_ * D_ * sizeof(bf16_t);
  bf16_t* wob  = (bf16_t*)w; w += (size_t)D_ * D_ * sizeof(bf16_t);
  bf16_t* qb   = (bf16_t*)w; w += nTok * sizeof(bf16_t);
  bf16_t* kbuf = (bf16_t*)w; w += nTok * sizeof(bf16_t);
  bf16_t* vT   = (bf16_t*)w; w += nTok * sizeof(bf16_t);
  bf16_t* ob   = xb;  // xb is dead after qkv_rope_gemm; reuse for attn output

  const int n1 = B_ * S_ * D_, n2 = 3 * D_ * D_, n3 = D_ * D_;
  cvt_f32_bf16<<<(n1 + 255) / 256, 256, 0, stream>>>(x, xb, n1);
  cvt_f32_bf16<<<(n2 + 255) / 256, 256, 0, stream>>>(qkv, qkvb, n2);
  cvt_f32_bf16<<<(n3 + 255) / 256, 256, 0, stream>>>(wo, wob, n3);

  // 48 tasks ({Q,K,V} x 16 heads) x 64 blocks of 8 M-tiles
  qkv_rope_gemm<<<dim3(48, 64), 256, 0, stream>>>(xb, qkvb, qb, kbuf, vT,
                                                  khOut, vhOut);
  // one wave per (b, h, 16-row q tile): 4*16*128 = 8192 waves
  attn_fwd<<<dim3(8192), 32, 0, stream>>>(qb, kbuf, vT, ob);
  // out projection: 16 column groups x 64 blocks of 8 M-tiles
  out_proj<<<dim3(16, 512 / 8), 256, 0, stream>>>(ob, wob, out);
}
